// Experts_PyTorch_2001454760318
// MI455X (gfx1250) — compile-verified, matrix operand order checked by reading, not run on hardware
//
#include <hip/hip_runtime.h>
#include <stdint.h>

// ---- CDNA5 vector / addrspace types -------------------------------------
typedef __attribute__((ext_vector_type(16))) __bf16       v16bf;
typedef __attribute__((ext_vector_type(8)))  float        v8f;
typedef __attribute__((ext_vector_type(4)))  unsigned int v4u;
typedef __attribute__((ext_vector_type(2)))  unsigned int v2u;
typedef __attribute__((ext_vector_type(4)))  int          v4i;

#define AS_GLOBAL __attribute__((address_space(1)))
#define AS_LDS    __attribute__((address_space(3)))

// ---- problem constants ---------------------------------------------------
#define NE    8
#define NT    32768
#define DIN   1024
#define DOUT  4096

// ---- tiling --------------------------------------------------------------
#define BM    128
#define BN    128
#define BK    32
#define KT    (DIN / BK)      // 32 K-steps
#define LDA   40              // padded LDS row stride (bf16): 80B rows, 16B aligned
#define THREADS 256

// ---- async-to-LDS availability (probe-safe) ------------------------------
#if defined(__has_builtin)
#  if __has_builtin(__builtin_amdgcn_global_load_async_to_lds_b128) && \
      __has_builtin(__builtin_amdgcn_s_wait_asynccnt)
#    define HAVE_ASYNC_LDS 1
#  endif
#endif
#ifndef HAVE_ASYNC_LDS
#  define HAVE_ASYNC_LDS 0
#endif

union Frag32 { v16bf v; v4u q[2]; };

// split fp32 into truncated-hi bf16 + RNE-rounded residual bf16
__device__ __forceinline__ void split_bf16(float x, unsigned int& hi, unsigned int& lo) {
    unsigned int b = __float_as_uint(x);
    hi = b >> 16;                                   // truncate -> exact residual
    float r = x - __uint_as_float(b & 0xffff0000u);
    unsigned int rb = __float_as_uint(r);
    lo = (rb + 0x7fffu + ((rb >> 16) & 1u)) >> 16;  // round-to-nearest-even
}

// 16B global -> LDS, async if available (ASYNCcnt), else sync load + ds_store
__device__ __forceinline__ void g2l_b128(const unsigned short* g, unsigned short* l) {
#if HAVE_ASYNC_LDS
    v4i* gp = (v4i*)(void*)g;   // generic typed ptr (C-style drops const)
    v4i* lp = (v4i*)(void*)l;
    __builtin_amdgcn_global_load_async_to_lds_b128(
        (AS_GLOBAL v4i*)gp, (AS_LDS v4i*)lp, 0, 0);
#else
    *(v4u*)l = *(const v4u*)g;
#endif
}

__device__ __forceinline__ void async_fence() {
#if HAVE_ASYNC_LDS
    __builtin_amdgcn_s_wait_asynccnt(0);
#endif
}

// =========================================================================
// pre-pass: split fp32 tensor into bf16 hi / lo planes (one-shot, bw-bound)
// =========================================================================
__global__ __launch_bounds__(256)
void split_planes(const float* __restrict__ src,
                  unsigned short* __restrict__ hi,
                  unsigned short* __restrict__ lo, int n4) {
    int i = blockIdx.x * blockDim.x + threadIdx.x;
    const int stride = gridDim.x * blockDim.x;
    for (; i < n4; i += stride) {
        float4 v = ((const float4*)src)[i];
        unsigned int h0, h1, h2, h3, l0, l1, l2, l3;
        split_bf16(v.x, h0, l0); split_bf16(v.y, h1, l1);
        split_bf16(v.z, h2, l2); split_bf16(v.w, h3, l3);
        v2u th; th.x = h0 | (h1 << 16); th.y = h2 | (h3 << 16);
        v2u tl; tl.x = l0 | (l1 << 16); tl.y = l2 | (l3 << 16);
        ((v2u*)hi)[i] = th;
        ((v2u*)lo)[i] = tl;
    }
}

// =========================================================================
// grouped-expert GEMM, bf16x3 split precision, WMMA 16x16x32
//   MODE 1: operands preconverted in d_ws, async-to-LDS ping-pong staging
//   MODE 0: fallback, converts fp32 inline (register double buffer)
// =========================================================================
template<int MODE>
__global__ __launch_bounds__(THREADS)
void moe_gemm(const float* __restrict__ x, const float* __restrict__ w,
              const unsigned short* __restrict__ xh, const unsigned short* __restrict__ xl,
              const unsigned short* __restrict__ wh, const unsigned short* __restrict__ wl,
              const float* __restrict__ bias, const long long* __restrict__ counts,
              float* __restrict__ out) {
    __shared__ unsigned short sAh[2][BM * LDA];
    __shared__ unsigned short sAl[2][BM * LDA];
    __shared__ unsigned short sBh[2][BN * LDA];
    __shared__ unsigned short sBl[2][BN * LDA];

    const int e  = blockIdx.x >> 10;      // 1024 tiles per expert
    const int t  = blockIdx.x & 1023;
    const int tm = t >> 5;
    const int tn = t & 31;

    long long rowOff = 0;
    for (int j = 0; j < e; ++j) rowOff += counts[j];

    const int tid  = threadIdx.x;
    const int wave = tid >> 5;
    const int lane = tid & 31;
    const int wm   = wave & 1;
    const int wn   = wave >> 1;
    const int lr   = lane & 15;
    const int lh   = lane >> 4;

    // ---- MODE 1 staging state: 2 x 16B chunks per plane per thread -------
    const size_t aOff = (size_t)(rowOff + (long long)tm * BM) * DIN;
    const size_t bOff = ((size_t)e * DOUT + (size_t)tn * BN) * (size_t)DIN;
    const unsigned short* aH = xh + aOff;
    const unsigned short* aL = xl + aOff;
    const unsigned short* bH = wh + bOff;
    const unsigned short* bL = wl + bOff;

    int cr[2], co[2];                 // chunk row / elem offset (8 bf16 = 16B)
#pragma unroll
    for (int i = 0; i < 2; ++i) {
        int c = tid + i * THREADS;    // 0..511 chunks per plane
        cr[i] = c >> 2;               // 4 chunks per 32-elem row
        co[i] = (c & 3) * 8;
    }

    auto stagePre = [&](int kt, int buf) {
        const size_t k0 = (size_t)kt * BK;
#pragma unroll
        for (int i = 0; i < 2; ++i) {
            const size_t go = (size_t)cr[i] * DIN + k0 + co[i];
            const int    lo_ = cr[i] * LDA + co[i];
            g2l_b128(aH + go, &sAh[buf][lo_]);
            g2l_b128(aL + go, &sAl[buf][lo_]);
            g2l_b128(bH + go, &sBh[buf][lo_]);
            g2l_b128(bL + go, &sBl[buf][lo_]);
        }
    };

    // ---- MODE 0 staging state: 4 float4 per thread per operand -----------
    const float* aBase = x + aOff;
    const float* bBase = w + bOff;
    int rows4[4], cols4[4];
#pragma unroll
    for (int i = 0; i < 4; ++i) {
        int g    = i * THREADS + tid;
        rows4[i] = g >> 3;
        cols4[i] = (g & 7) * 4;
    }
    float4 aReg[4], bReg[4];

    auto loadRegs = [&](int kt) {
        const float* ap = aBase + kt * BK;
        const float* bp = bBase + kt * BK;
#pragma unroll
        for (int i = 0; i < 4; ++i) {
            aReg[i] = *(const float4*)(ap + (size_t)rows4[i] * DIN + cols4[i]);
            bReg[i] = *(const float4*)(bp + (size_t)rows4[i] * DIN + cols4[i]);
        }
    };
    auto storeRegs = [&](int buf) {
#pragma unroll
        for (int i = 0; i < 4; ++i) {
            const int idx = rows4[i] * LDA + cols4[i];
            unsigned int h0, h1, h2, h3, l0, l1, l2, l3;
            const float* av = (const float*)&aReg[i];
            split_bf16(av[0], h0, l0); split_bf16(av[1], h1, l1);
            split_bf16(av[2], h2, l2); split_bf16(av[3], h3, l3);
            v2u th; th.x = h0 | (h1 << 16); th.y = h2 | (h3 << 16);
            v2u tl; tl.x = l0 | (l1 << 16); tl.y = l2 | (l3 << 16);
            *(v2u*)&sAh[buf][idx] = th;
            *(v2u*)&sAl[buf][idx] = tl;
            const float* bv = (const float*)&bReg[i];
            split_bf16(bv[0], h0, l0); split_bf16(bv[1], h1, l1);
            split_bf16(bv[2], h2, l2); split_bf16(bv[3], h3, l3);
            th.x = h0 | (h1 << 16); th.y = h2 | (h3 << 16);
            tl.x = l0 | (l1 << 16); tl.y = l2 | (l3 << 16);
            *(v2u*)&sBh[buf][idx] = th;
            *(v2u*)&sBl[buf][idx] = tl;
        }
    };

    // ---- accumulators ----------------------------------------------------
    v8f acc[4][2];
    const v8f vzero = {0.f, 0.f, 0.f, 0.f, 0.f, 0.f, 0.f, 0.f};
#pragma unroll
    for (int mi = 0; mi < 4; ++mi)
#pragma unroll
        for (int ni = 0; ni < 2; ++ni) acc[mi][ni] = vzero;

    auto compute = [&](int buf) {
        Frag32 ah[4], al[4], bh[2], bl[2];
        const unsigned short* pAh = sAh[buf];
        const unsigned short* pAl = sAl[buf];
        const unsigned short* pBh = sBh[buf];
        const unsigned short* pBl = sBl[buf];
#pragma unroll
        for (int mi = 0; mi < 4; ++mi) {
            const int row = wm * 64 + mi * 16 + lr;
            const int o0  = row * LDA + lh * 8;
            ah[mi].q[0] = *(const v4u*)&pAh[o0]; ah[mi].q[1] = *(const v4u*)&pAh[o0 + 16];
            al[mi].q[0] = *(const v4u*)&pAl[o0]; al[mi].q[1] = *(const v4u*)&pAl[o0 + 16];
        }
#pragma unroll
        for (int ni = 0; ni < 2; ++ni) {
            const int col = wn * 32 + ni * 16 + lr;
            const int o0  = col * LDA + lh * 8;
            bh[ni].q[0] = *(const v4u*)&pBh[o0]; bh[ni].q[1] = *(const v4u*)&pBh[o0 + 16];
            bl[ni].q[0] = *(const v4u*)&pBl[o0]; bl[ni].q[1] = *(const v4u*)&pBl[o0 + 16];
        }
#pragma unroll
        for (int mi = 0; mi < 4; ++mi) {
#pragma unroll
            for (int ni = 0; ni < 2; ++ni) {
                v8f c = acc[mi][ni];
                c = __builtin_amdgcn_wmma_f32_16x16x32_bf16(
                        false, ah[mi].v, false, bh[ni].v, (short)0, c, false, false);
                c = __builtin_amdgcn_wmma_f32_16x16x32_bf16(
                        false, ah[mi].v, false, bl[ni].v, (short)0, c, false, false);
                c = __builtin_amdgcn_wmma_f32_16x16x32_bf16(
                        false, al[mi].v, false, bh[ni].v, (short)0, c, false, false);
                acc[mi][ni] = c;
            }
        }
    };

    // ---- software pipeline: 1 barrier / iteration, LDS ping-pong ---------
    if constexpr (MODE == 1) {
        stagePre(0, 0);
        async_fence();
    } else {
        loadRegs(0);
        storeRegs(0);
    }
    __syncthreads();

    for (int kt = 0; kt < KT; ++kt) {
        const int cur = kt & 1, nxt = cur ^ 1;
        if constexpr (MODE == 1) {
            if (kt + 1 < KT) stagePre(kt + 1, nxt);   // async fill next buffer
        } else {
            if (kt + 1 < KT) loadRegs(kt + 1);        // global loads in flight
            if (kt + 2 < KT) {
                __builtin_prefetch(aBase + (size_t)(kt + 2) * BK + (size_t)rows4[0] * DIN, 0, 1);
                __builtin_prefetch(bBase + (size_t)(kt + 2) * BK + (size_t)rows4[0] * DIN, 0, 1);
            }
        }

        compute(cur);

        if constexpr (MODE == 1) {
            async_fence();                            // own async ops landed
        } else {
            if (kt + 1 < KT) storeRegs(nxt);
        }
        __syncthreads();                              // everyone's tile ready
    }

    // ---- epilogue: C layout VGPR r -> M=r / r+8 by lane half, N=lane%16 --
    const long long rowBase = rowOff + (long long)tm * BM + wm * 64;
    const int       colBase = tn * BN + wn * 32;
#pragma unroll
    for (int ni = 0; ni < 2; ++ni) {
        const int gn = colBase + ni * 16 + lr;
        const float bv = bias[e * DOUT + gn];
#pragma unroll
        for (int mi = 0; mi < 4; ++mi) {
#pragma unroll
            for (int r = 0; r < 8; ++r) {
                const long long gm = rowBase + mi * 16 + lh * 8 + r;
                out[gm * (long long)DOUT + gn] = acc[mi][ni][r] + bv;
            }
        }
    }
}

extern "C" void kernel_launch(void* const* d_in, const int* in_sizes, int n_in,
                              void* d_out, int out_size, void* d_ws, size_t ws_size,
                              hipStream_t stream) {
    const float*     x   = (const float*)d_in[0];
    const float*     w   = (const float*)d_in[1];
    const float*     b   = (const float*)d_in[2];
    const long long* cnt = (const long long*)d_in[3];
    float*           out = (float*)d_out;

    const int tilesPerExpert = ((NT / NE) / BM) * (DOUT / BN);  // 1024
    dim3 grid(NE * tilesPerExpert);                             // 8192 blocks

    const size_t planeX = (size_t)NT * DIN;          // elements
    const size_t planeW = (size_t)NE * DOUT * DIN;
    const size_t need   = (planeX + planeW) * 2u * sizeof(unsigned short);

    if (ws_size >= need) {
        unsigned short* xh = (unsigned short*)d_ws;
        unsigned short* xl = xh + planeX;
        unsigned short* wh = xl + planeX;
        unsigned short* wl = wh + planeW;
        split_planes<<<2048, 256, 0, stream>>>(x, xh, xl, (int)(planeX / 4));
        split_planes<<<2048, 256, 0, stream>>>(w, wh, wl, (int)(planeW / 4));
        moe_gemm<1><<<grid, THREADS, 0, stream>>>(x, w, xh, xl, wh, wl, b, cnt, out);
    } else {
        moe_gemm<0><<<grid, THREADS, 0, stream>>>(x, w, nullptr, nullptr,
                                                  nullptr, nullptr, b, cnt, out);
    }
}
